// SwinTransformerV2TinyBackbone_14894946583370
// MI455X (gfx1250) — compile-verified, matrix operand order checked by reading, not run on hardware
//
#include <hip/hip_runtime.h>
#include <hip/hip_bf16.h>
#include <math.h>

typedef __attribute__((ext_vector_type(16))) __bf16 v16bf;
typedef __attribute__((ext_vector_type(8)))  float  v8f;

#define WMMA_BF16(a,b,c) __builtin_amdgcn_wmma_f32_16x16x32_bf16(false,(a),false,(b),(short)0,(c),false,false)

// gfx1250 async global->LDS copy (16B per lane), tracked by ASYNCcnt.
#define ASYNC_LDS_B128(ldsOff, gaddr) \
  asm volatile("global_load_async_to_lds_b128 %0, %1, off" :: "v"(ldsOff), "v"(gaddr) : "memory")
#define WAIT_ASYNC_0() asm volatile("s_wait_asynccnt 0x0" ::: "memory")
#define WAIT_ASYNC_2() asm volatile("s_wait_asynccnt 0x2" ::: "memory")

__device__ __forceinline__ unsigned short bf16bits(float x) {
  return __builtin_bit_cast(unsigned short, (__bf16)x);
}
__device__ __forceinline__ unsigned pk2(float a, float b) {
  return (unsigned)bf16bits(a) | ((unsigned)bf16bits(b) << 16);
}
__device__ __forceinline__ unsigned lds_off(const void* p) {
  return (unsigned)(unsigned long long)p;
}

union FragU { uint4 u[2]; v16bf v; };
__device__ __forceinline__ v16bf load_frag(const __bf16* p0, const __bf16* p1) {
  FragU f;
  f.u[0] = *(const uint4*)p0;   // ds_load_b128
  f.u[1] = *(const uint4*)p1;   // ds_load_b128
  return f.v;
}

// ---------------------------------------------------------------------------
// GEMM: C[M,N] = A[M,K] * W[N,K]^T (+bias)(+GELU). A,W bf16; K%32==0, M%64==0.
// 256 thr = 8 waves, 64x64 tile, 2 WMMA accumulators/wave, double-buffered
// async global->LDS pipeline.
// ---------------------------------------------------------------------------
__global__ __launch_bounds__(256) void gemm_bf16_kernel(
    const __bf16* __restrict__ A, const __bf16* __restrict__ Wt,
    const float* __restrict__ bias, float* __restrict__ outF,
    __bf16* __restrict__ outB, int M, int N, int K, int doGelu)
{
  __shared__ __bf16 As[2][64][32];   // [buf][m][k]
  __shared__ __bf16 Bs[2][64][32];   // [buf][n][k]
  const int tid = threadIdx.x, lane = tid & 31, wv = tid >> 5;
  const int m0 = blockIdx.x * 64, n0 = blockIdx.y * 64;
  const int tm  = (wv & 3) * 16;
  const int tn0 = (wv >> 2) * 32;
  const int hl = lane >> 4, rr = lane & 15;

  // fill assignment: thread -> row fm (0..63), k-chunk fk (0/8/16/24) = 16B
  const int fm = tid >> 2;
  const int fk = (tid & 3) * 8;
  const int nrow = (n0 + fm < N) ? (n0 + fm) : (N - 1);   // clamp: junk cols never stored
  const __bf16* Ap = A  + (size_t)(m0 + fm) * K + fk;
  const __bf16* Bp = Wt + (size_t)nrow * K + fk;
  const unsigned ldsA0 = lds_off(&As[0][fm][fk]), ldsA1 = lds_off(&As[1][fm][fk]);
  const unsigned ldsB0 = lds_off(&Bs[0][fm][fk]), ldsB1 = lds_off(&Bs[1][fm][fk]);

  v8f acc0 = {}, acc1 = {};

  auto issue = [&](int k0, int bi) {
    ASYNC_LDS_B128(bi ? ldsA1 : ldsA0, (unsigned long long)(const void*)(Ap + k0));
    ASYNC_LDS_B128(bi ? ldsB1 : ldsB0, (unsigned long long)(const void*)(Bp + k0));
  };
  auto compute = [&](int bi) {
    v16bf af  = load_frag(&As[bi][tm + rr][hl * 8],       &As[bi][tm + rr][16 + hl * 8]);
    v16bf bf0 = load_frag(&Bs[bi][tn0 + rr][hl * 8],      &Bs[bi][tn0 + rr][16 + hl * 8]);
    v16bf bf1 = load_frag(&Bs[bi][tn0 + 16 + rr][hl * 8], &Bs[bi][tn0 + 16 + rr][16 + hl * 8]);
    acc0 = WMMA_BF16(af, bf0, acc0);
    acc1 = WMMA_BF16(af, bf1, acc1);
  };

  issue(0, 0);
  int cur = 0, k0 = 0;
  for (; k0 + 32 < K; k0 += 32) {
    issue(k0 + 32, cur ^ 1);   // prefetch next panel into alternate buffer
    WAIT_ASYNC_2();            // wait for current panel only
    __syncthreads();
    compute(cur);
    __syncthreads();
    cur ^= 1;
  }
  WAIT_ASYNC_0();
  __syncthreads();
  compute(cur);

  const int col0 = n0 + tn0 + rr;
  const int col1 = col0 + 16;
  float bs0 = 0.f, bs1 = 0.f;
  if (bias) {                      // clamped loads: junk values never stored
    bs0 = bias[col0 < N ? col0 : N - 1];
    bs1 = bias[col1 < N ? col1 : N - 1];
  }
#pragma unroll
  for (int i = 0; i < 8; ++i) {
    int row = m0 + tm + i + 8 * hl;
    if (col0 < N) {
      float v = acc0[i] + bs0;
      if (doGelu) v = 0.5f * v * (1.f + erff(v * 0.70710678f));
      if (outF) outF[(size_t)row * N + col0] = v;
      if (outB) outB[(size_t)row * N + col0] = (__bf16)v;
    }
    if (col1 < N) {
      float v = acc1[i] + bs1;
      if (doGelu) v = 0.5f * v * (1.f + erff(v * 0.70710678f));
      if (outF) outF[(size_t)row * N + col1] = v;
      if (outB) outB[(size_t)row * N + col1] = (__bf16)v;
    }
  }
}

// ---------------------------------------------------------------------------
// Weight f32 -> bf16 (packed pair stores), optional zero-pad Ks -> Kd
// ---------------------------------------------------------------------------
__global__ void conv_w_kernel(const float* __restrict__ src, __bf16* __restrict__ dst,
                              int rows, int Ks, int Kd)
{
  size_t i = (size_t)blockIdx.x * blockDim.x + threadIdx.x;
  size_t tot = (size_t)rows * (Kd >> 1);
  if (i >= tot) return;
  int kp = (int)(i % (size_t)(Kd >> 1));
  size_t row = i / (size_t)(Kd >> 1);
  int k = kp * 2;
  float a = (k     < Ks) ? src[row * Ks + k]     : 0.f;
  float b = (k + 1 < Ks) ? src[row * Ks + k + 1] : 0.f;
  ((unsigned*)dst)[i] = pk2(a, b);
}

// ---------------------------------------------------------------------------
// CPB bias table: raw[d*heads+h]
// ---------------------------------------------------------------------------
__global__ void cpb_kernel(const float* __restrict__ w1, const float* __restrict__ b1,
                           const float* __restrict__ w2, float* __restrict__ raw, int heads)
{
  int i = blockIdx.x * blockDim.x + threadIdx.x;
  if (i >= 225 * heads) return;
  int d = i / heads, h = i % heads;
  int di = d / 15, dj = d % 15;
  float v0 = (di - 7) * (8.f / 7.f);
  float v1 = (dj - 7) * (8.f / 7.f);
  float t0 = (v0 >= 0.f ? 1.f : -1.f) * log2f(fabsf(v0) + 1.f) * (1.f / 3.f);
  float t1 = (v1 >= 0.f ? 1.f : -1.f) * log2f(fabsf(v1) + 1.f) * (1.f / 3.f);
  float acc = 0.f;
  for (int j = 0; j < 512; ++j) {
    float hh = fmaxf(t0 * w1[2 * j] + t1 * w1[2 * j + 1] + b1[j], 0.f);
    acc += hh * w2[h * 512 + j];
  }
  raw[d * heads + h] = acc;
}

// ---------------------------------------------------------------------------
// Fused window attention: one block per (window, head). 64 tokens, hd=32.
// q/k gathered via async global->LDS; v converted to bf16 (transposed).
// ---------------------------------------------------------------------------
__device__ __forceinline__ int region_id(int p, int L, int shift) {
  return (p < L - 8) ? 0 : ((p < L - shift) ? 1 : 2);
}

__global__ __launch_bounds__(256) void attn_kernel(
    const float* __restrict__ qkv, __bf16* __restrict__ out,
    const float* __restrict__ logit_scale, const float* __restrict__ biasRaw,
    int B, int H, int Wd, int C, int heads, int shift)
{
  __shared__ float  qf[64][32];
  __shared__ float  kf[64][32];
  __shared__ __bf16 qb[64][32];
  __shared__ __bf16 kb[64][32];
  __shared__ __bf16 vb[32][64];   // [d][token]
  __shared__ float  Sf[64][64];
  __shared__ __bf16 Pb[64][64];
  __shared__ int    gidx[64];

  const int h   = blockIdx.x % heads;
  const int wid = blockIdx.x / heads;
  const int nW  = Wd / 8, nH = H / 8;
  const int b   = wid / (nH * nW);
  const int wr  = (wid / nW) % nH;
  const int wc  = wid % nW;
  const int tid = threadIdx.x, lane = tid & 31, wv = tid >> 5;
  const int hl  = lane >> 4, rr = lane & 15;
  const int C3  = 3 * C;

  if (tid < 64) {
    int r = tid >> 3, c = tid & 7;
    int ir = (wr * 8 + r + shift) % H;
    int ic = (wc * 8 + c + shift) % Wd;
    gidx[tid] = (b * H + ir) * Wd + ic;
  }
  __syncthreads();

  // gather: 512 float4-chunks per matrix, 2 per thread.
  // q/k: async copy straight into LDS; v: convert to bf16, transposed.
#pragma unroll
  for (int it = 0; it < 2; ++it) {
    int ch = tid + it * 256;            // 0..511
    int n = ch >> 3, p = ch & 7;        // token, float4 index
    const float* base = qkv + (size_t)gidx[n] * C3 + h * 32 + p * 4;
    ASYNC_LDS_B128(lds_off(&qf[n][p * 4]), (unsigned long long)(const void*)(base));
    ASYNC_LDS_B128(lds_off(&kf[n][p * 4]), (unsigned long long)(const void*)(base + C));
    float4 v4 = *(const float4*)(base + 2 * C);
    int d = p * 4;
    vb[d + 0][n] = (__bf16)v4.x;
    vb[d + 1][n] = (__bf16)v4.y;
    vb[d + 2][n] = (__bf16)v4.z;
    vb[d + 3][n] = (__bf16)v4.w;
  }
  WAIT_ASYNC_0();
  __syncthreads();

  // wave-parallel row normalization
#pragma unroll
  for (int i = 0; i < 8; ++i) {
    int row = wv * 8 + i;
    float qv = qf[row][lane];
    float s = qv * qv;
    for (int o = 16; o; o >>= 1) s += __shfl_xor(s, o, 32);
    qb[row][lane] = (__bf16)(qv / fmaxf(sqrtf(s), 1e-12f));
    float kv = kf[row][lane];
    s = kv * kv;
    for (int o = 16; o; o >>= 1) s += __shfl_xor(s, o, 32);
    kb[row][lane] = (__bf16)(kv / fmaxf(sqrtf(s), 1e-12f));
  }
  __syncthreads();

  const float scale = __expf(fminf(logit_scale[h], 4.60517019f)); // ln(100)

  // S = qn @ kn^T : 16 tiles, 2 per wave
#pragma unroll
  for (int t = 0; t < 2; ++t) {
    int ti = wv * 2 + t;
    int tm = (ti >> 2) * 16, tn = (ti & 3) * 16;
    v16bf af = load_frag(&qb[tm + rr][hl * 8], &qb[tm + rr][16 + hl * 8]);
    v16bf bf = load_frag(&kb[tn + rr][hl * 8], &kb[tn + rr][16 + hl * 8]);
    v8f acc = {};
    acc = WMMA_BF16(af, bf, acc);
#pragma unroll
    for (int i = 0; i < 8; ++i) {
      int row = tm + i + 8 * hl;   // query token
      int col = tn + rr;           // key token
      int rn = row >> 3, cn = row & 7, rm = col >> 3, cm = col & 7;
      int dIdx = (rn - rm + 7) * 15 + (cn - cm + 7);
      float braw = biasRaw[dIdx * heads + h];
      float v = acc[i] * scale + 16.f / (1.f + __expf(-braw));
      if (shift) {
        int gn = region_id(wr * 8 + rn, H, shift) * 3 + region_id(wc * 8 + cn, Wd, shift);
        int gm = region_id(wr * 8 + rm, H, shift) * 3 + region_id(wc * 8 + cm, Wd, shift);
        if (gn != gm) v -= 100.f;
      }
      Sf[row][col] = v;
    }
  }
  __syncthreads();

  // wave-parallel softmax: wave handles 8 rows; lane covers cols {lane, lane+32}
#pragma unroll
  for (int i = 0; i < 8; ++i) {
    int row = wv * 8 + i;
    float a = Sf[row][lane];
    float c2 = Sf[row][lane + 32];
    float mx = fmaxf(a, c2);
    for (int o = 16; o; o >>= 1) mx = fmaxf(mx, __shfl_xor(mx, o, 32));
    float ea = __expf(a - mx), eb = __expf(c2 - mx);
    float s = ea + eb;
    for (int o = 16; o; o >>= 1) s += __shfl_xor(s, o, 32);
    float inv = 1.f / s;
    Pb[row][lane]      = (__bf16)(ea * inv);
    Pb[row][lane + 32] = (__bf16)(eb * inv);
  }
  __syncthreads();

  // O = P @ V : 64x32 output, 8 tiles (one per wave), K=64 -> 2 WMMA steps
  {
    int tm = (wv & 3) * 16, tn = (wv >> 2) * 16;
    v8f acc = {};
#pragma unroll
    for (int k0 = 0; k0 < 64; k0 += 32) {
      v16bf af = load_frag(&Pb[tm + rr][k0 + hl * 8], &Pb[tm + rr][k0 + 16 + hl * 8]);
      v16bf bf = load_frag(&vb[tn + rr][k0 + hl * 8], &vb[tn + rr][k0 + 16 + hl * 8]);
      acc = WMMA_BF16(af, bf, acc);
    }
#pragma unroll
    for (int i = 0; i < 8; ++i) {
      int row = tm + i + 8 * hl;   // token
      int col = tn + rr;           // head-dim
      out[(size_t)gidx[row] * C + h * 32 + col] = (__bf16)acc[i];
    }
  }
}

// ---------------------------------------------------------------------------
// Residual + LayerNorm (wave per row). Updates x (f32) and optional bf16 copy.
// ---------------------------------------------------------------------------
__global__ __launch_bounds__(256) void add_ln_kernel(
    float* __restrict__ x, const float* __restrict__ hbuf,
    const float* __restrict__ g, const float* __restrict__ bta,
    __bf16* __restrict__ outB, int T, int C)
{
  int wv = threadIdx.x >> 5, lane = threadIdx.x & 31;
  int t = blockIdx.x * 8 + wv;
  if (t >= T) return;
  const float* hr = hbuf + (size_t)t * C;
  float* xr = x + (size_t)t * C;
  float hv[24];
  int nc = C >> 5;
  float s = 0.f;
  for (int i = 0; i < nc; ++i) { hv[i] = hr[i * 32 + lane]; s += hv[i]; }
  for (int o = 16; o; o >>= 1) s += __shfl_xor(s, o, 32);
  float m = s / C;
  float vv = 0.f;
  for (int i = 0; i < nc; ++i) { float d = hv[i] - m; vv += d * d; }
  for (int o = 16; o; o >>= 1) vv += __shfl_xor(vv, o, 32);
  float rinv = rsqrtf(vv / C + 1e-5f);
  for (int i = 0; i < nc; ++i) {
    int c = i * 32 + lane;
    float nv = xr[c] + (hv[i] - m) * rinv * g[c] + bta[c];
    xr[c] = nv;
    if (outB) outB[(size_t)t * C + c] = (__bf16)nv;
  }
}

__global__ __launch_bounds__(256) void ln_kernel(
    const float* __restrict__ in, float* __restrict__ out,
    const float* __restrict__ g, const float* __restrict__ bta,
    __bf16* __restrict__ outB, int T, int C)
{
  int wv = threadIdx.x >> 5, lane = threadIdx.x & 31;
  int t = blockIdx.x * 8 + wv;
  if (t >= T) return;
  const float* hr = in + (size_t)t * C;
  float* xr = out + (size_t)t * C;
  float hv[24];
  int nc = C >> 5;
  float s = 0.f;
  for (int i = 0; i < nc; ++i) { hv[i] = hr[i * 32 + lane]; s += hv[i]; }
  for (int o = 16; o; o >>= 1) s += __shfl_xor(s, o, 32);
  float m = s / C;
  float vv = 0.f;
  for (int i = 0; i < nc; ++i) { float d = hv[i] - m; vv += d * d; }
  for (int o = 16; o; o >>= 1) vv += __shfl_xor(vv, o, 32);
  float rinv = rsqrtf(vv / C + 1e-5f);
  for (int i = 0; i < nc; ++i) {
    int c = i * 32 + lane;
    float nv = (hv[i] - m) * rinv * g[c] + bta[c];
    xr[c] = nv;
    if (outB) outB[(size_t)t * C + c] = (__bf16)nv;
  }
}

// ---------------------------------------------------------------------------
// Gathers (emit bf16 GEMM inputs) / final transpose
// ---------------------------------------------------------------------------
// [B,3,256,256] -> [B,64,64,64] bf16, channels 48..63 zero-padded
__global__ void patch_gather_kernel(const float* __restrict__ x, __bf16* __restrict__ y, int B)
{
  size_t i = (size_t)blockIdx.x * blockDim.x + threadIdx.x;
  size_t total = (size_t)B * 64 * 64 * 64;
  if (i >= total) return;
  int c64 = i % 64;
  size_t t = i / 64;
  int wp = t % 64, hp = (t / 64) % 64, b = t / 4096;
  float v = 0.f;
  if (c64 < 48) {
    int c = c64 / 16, rem = c64 % 16, pi = rem / 4, pj = rem % 4;
    v = x[(((size_t)b * 3 + c) * 256 + hp * 4 + pi) * 256 + wp * 4 + pj];
  }
  y[i] = (__bf16)v;
}

// [B,H,W,C] f32 -> [B,H/2,W/2,4C] bf16 (x0|x1|x2|x3)
__global__ void merge_gather_kernel(const float* __restrict__ x, __bf16* __restrict__ y,
                                    int B, int H, int C)
{
  size_t i = (size_t)blockIdx.x * blockDim.x + threadIdx.x;
  int W2 = H / 2;
  size_t total = (size_t)B * W2 * W2 * 4 * C;
  if (i >= total) return;
  int c4 = i % (4 * C);
  size_t t = i / (4 * C);
  int w2 = t % W2, h2 = (t / W2) % W2, b = t / ((size_t)W2 * W2);
  int q = c4 / C, cc = c4 % C;
  int sh = 2 * h2 + (q & 1);
  int sw = 2 * w2 + (q >> 1);
  y[i] = (__bf16)x[(((size_t)b * H + sh) * H + sw) * C + cc];
}

__global__ void transpose_kernel(const float* __restrict__ in, float* __restrict__ out,
                                 int B, int H, int C)
{
  size_t i = (size_t)blockIdx.x * blockDim.x + threadIdx.x;
  size_t total = (size_t)B * C * H * H;
  if (i >= total) return;
  int w = i % H;
  int hh = (i / H) % H;
  int c = (i / ((size_t)H * H)) % C;
  int b = i / ((size_t)C * H * H);
  out[i] = in[(((size_t)b * H + hh) * H + w) * C + c];
}

// ---------------------------------------------------------------------------
// Host orchestration
// ---------------------------------------------------------------------------
static void launch_gemm(const __bf16* A, const __bf16* Wt, const float* bias,
                        float* outF, __bf16* outB, int M, int N, int K, int gelu,
                        hipStream_t s)
{
  dim3 grid((M + 63) / 64, (N + 63) / 64);
  hipLaunchKernelGGL(gemm_bf16_kernel, grid, dim3(256), 0, s, A, Wt, bias, outF, outB, M, N, K, gelu);
}

static void launch_conv_w(const float* src, __bf16* dst, int rows, int Ks, int Kd,
                          hipStream_t s)
{
  size_t tot = (size_t)rows * (Kd / 2);
  hipLaunchKernelGGL(conv_w_kernel, dim3((tot + 255) / 256), dim3(256), 0, s,
                     src, dst, rows, Ks, Kd);
}

extern "C" void kernel_launch(void* const* d_in, const int* in_sizes, int n_in,
                              void* d_out, int out_size, void* d_ws, size_t ws_size,
                              hipStream_t stream)
{
  (void)n_in; (void)out_size; (void)ws_size;
  auto P = [&](int i) { return (const float*)d_in[i]; };
  const float* x_in = P(0);
  const int B = in_sizes[0] / (3 * 256 * 256);

  char* ws = (char*)d_ws;
  float*  bufX    = (float*) (ws);                  // f32 activations     (24 MiB)
  __bf16* bufXb   = (__bf16*)(ws + (24ull  << 20)); // bf16 copy of x      (12 MiB)
  float*  bufA    = (float*) (ws + (36ull  << 20)); // f32 tmp (qkv etc.)  (76 MiB)
  __bf16* bufHb   = (__bf16*)(ws + (112ull << 20)); // bf16 GEMM-A tmp     (52 MiB)
  __bf16* bufWb   = (__bf16*)(ws + (164ull << 20)); // bf16 weights        ( 8 MiB)
  float*  bufBias = (float*) (ws + (172ull << 20)); // CPB table

  static const int DEPTHS[4] = {2, 2, 6, 2};
  static const int HEADS_[4] = {3, 6, 12, 24};

  // ---- patch embed: gather(->bf16, K pad 48->64) -> GEMM -> LN ----
  {
    int T0 = B * 64 * 64;
    size_t tot = (size_t)T0 * 64;
    hipLaunchKernelGGL(patch_gather_kernel, dim3((tot + 255) / 256), dim3(256), 0, stream,
                       x_in, bufHb, B);
    launch_conv_w(P(1), bufWb, 96, 48, 64, stream);
    launch_gemm(bufHb, bufWb, P(2), bufA, nullptr, T0, 96, 64, 0, stream);
    hipLaunchKernelGGL(ln_kernel, dim3((T0 + 7) / 8), dim3(256), 0, stream,
                       bufA, bufX, P(3), P(4), bufXb, T0, 96);
  }

  int cur = 5;
  int Hs = 64, Cs = 96;
  for (int s = 0; s < 4; ++s) {
    const int heads = HEADS_[s];
    const int H = Hs;
    const int T = B * H * H;
    const int shift_sz = (8 >= H) ? 0 : 4;
    const int nWin = B * (H / 8) * (H / 8);

    for (int bl = 0; bl < DEPTHS[s]; ++bl) {
      const float* qkv_w = P(cur + 0);
      const float* qkv_b = P(cur + 1);
      const float* proj_w = P(cur + 2);
      const float* proj_b = P(cur + 3);
      const float* logit_scale = P(cur + 4);
      const float* cpb_w1 = P(cur + 5);
      const float* cpb_b1 = P(cur + 6);
      const float* cpb_w2 = P(cur + 7);
      const float* n1_g = P(cur + 8);
      const float* n1_b = P(cur + 9);
      const float* n2_g = P(cur + 10);
      const float* n2_b = P(cur + 11);
      const float* m_w1 = P(cur + 12);
      const float* m_b1 = P(cur + 13);
      const float* m_w2 = P(cur + 14);
      const float* m_b2 = P(cur + 15);
      cur += 16;
      const int sh = (bl & 1) ? shift_sz : 0;

      hipLaunchKernelGGL(cpb_kernel, dim3((225 * heads + 255) / 256), dim3(256), 0, stream,
                         cpb_w1, cpb_b1, cpb_w2, bufBias, heads);
      // qkv: x(bf16) @ Wqkv -> f32
      launch_conv_w(qkv_w, bufWb, 3 * Cs, Cs, Cs, stream);
      launch_gemm(bufXb, bufWb, qkv_b, bufA, nullptr, T, 3 * Cs, Cs, 0, stream);
      // attention -> bf16
      hipLaunchKernelGGL(attn_kernel, dim3(nWin * heads), dim3(256), 0, stream,
                         bufA, bufHb, logit_scale, bufBias, B, H, H, Cs, heads, sh);
      // proj: attn(bf16) @ Wproj -> f32
      launch_conv_w(proj_w, bufWb, Cs, Cs, Cs, stream);
      launch_gemm(bufHb, bufWb, proj_b, bufA, nullptr, T, Cs, Cs, 0, stream);
      hipLaunchKernelGGL(add_ln_kernel, dim3((T + 7) / 8), dim3(256), 0, stream,
                         bufX, bufA, n1_g, n1_b, bufXb, T, Cs);
      // MLP1: x(bf16) @ W1 -> GELU -> bf16 only
      launch_conv_w(m_w1, bufWb, 4 * Cs, Cs, Cs, stream);
      launch_gemm(bufXb, bufWb, m_b1, nullptr, bufHb, T, 4 * Cs, Cs, 1, stream);
      // MLP2: h(bf16) @ W2 -> f32
      launch_conv_w(m_w2, bufWb, Cs, 4 * Cs, 4 * Cs, stream);
      launch_gemm(bufHb, bufWb, m_b2, bufA, nullptr, T, Cs, 4 * Cs, 0, stream);
      hipLaunchKernelGGL(add_ln_kernel, dim3((T + 7) / 8), dim3(256), 0, stream,
                         bufX, bufA, n2_g, n2_b, bufXb, T, Cs);
    }

    if (s < 3) {
      const float* red_w = P(cur + 0);
      const float* mn_g = P(cur + 1);
      const float* mn_b = P(cur + 2);
      cur += 3;
      int T2 = T / 4;
      size_t tot = (size_t)T2 * 4 * Cs;
      hipLaunchKernelGGL(merge_gather_kernel, dim3((tot + 255) / 256), dim3(256), 0, stream,
                         bufX, bufHb, B, H, Cs);
      launch_conv_w(red_w, bufWb, 2 * Cs, 4 * Cs, 4 * Cs, stream);
      launch_gemm(bufHb, bufWb, nullptr, bufA, nullptr, T2, 2 * Cs, 4 * Cs, 0, stream);
      hipLaunchKernelGGL(ln_kernel, dim3((T2 + 7) / 8), dim3(256), 0, stream,
                         bufA, bufX, mn_g, mn_b, bufXb, T2, 2 * Cs);
      Hs /= 2;
      Cs *= 2;
    }
  }

  // ---- final norm + transpose to [B,768,8,8] ----
  {
    int T = B * 8 * 8;
    hipLaunchKernelGGL(ln_kernel, dim3((T + 7) / 8), dim3(256), 0, stream,
                       bufX, bufA, P(cur), P(cur + 1), nullptr, T, 768);
    size_t tot = (size_t)B * 768 * 64;
    hipLaunchKernelGGL(transpose_kernel, dim3((tot + 255) / 256), dim3(256), 0, stream,
                       bufA, (float*)d_out, B, 8, 768);
  }
}
